// TinyTopkRouter_18923625906578
// MI455X (gfx1250) — compile-verified, matrix operand order checked by reading, not run on hardware
//
#include <hip/hip_runtime.h>
#include <hip/hip_bf16.h>

typedef __attribute__((ext_vector_type(16))) __bf16 v16bf;
typedef __attribute__((ext_vector_type(8)))  float  v8f;
typedef __attribute__((ext_vector_type(4)))  float  f32x4;

#define HIDDEN 4096
#define NEXP   64
#define TOPK   6
#define TOKENS 16384
#define SCALE  2.5f

// Load 16 fp32 covering K offsets {0..7, 16..23} from p, convert to packed bf16
// fragment matching the CDNA5 16-bit A/B WMMA layout (this lane's half-K slice).
__device__ __forceinline__ v16bf load16_cvt(const float* __restrict__ p) {
    const f32x4* q = (const f32x4*)p;
    f32x4 f0 = q[0];   // K+0..3
    f32x4 f1 = q[1];   // K+4..7
    f32x4 f2 = q[4];   // K+16..19
    f32x4 f3 = q[5];   // K+20..23
    v16bf r;
    r[0]  = (__bf16)f0[0]; r[1]  = (__bf16)f0[1]; r[2]  = (__bf16)f0[2]; r[3]  = (__bf16)f0[3];
    r[4]  = (__bf16)f1[0]; r[5]  = (__bf16)f1[1]; r[6]  = (__bf16)f1[2]; r[7]  = (__bf16)f1[3];
    r[8]  = (__bf16)f2[0]; r[9]  = (__bf16)f2[1]; r[10] = (__bf16)f2[2]; r[11] = (__bf16)f2[3];
    r[12] = (__bf16)f3[0]; r[13] = (__bf16)f3[1]; r[14] = (__bf16)f3[2]; r[15] = (__bf16)f3[3];
    return r;
}

__global__ __launch_bounds__(128)
void TinyTopkRouter_kernel(const float* __restrict__ x,
                           const float* __restrict__ w,
                           int*   __restrict__ idx_out,
                           float* __restrict__ w_out) {
    __shared__ float sLogits[4][32][NEXP];   // 32 KB: per-wave 32 tokens x 64 experts

    const int lane = threadIdx.x & 31;
    const int wave = threadIdx.x >> 5;
    const int tokenBase = (blockIdx.x * 4 + wave) * 32;   // 32 tokens per wave

    // A/B fragment addressing: lanes 0-15 hold row (lane), K slice {0..7,16..23};
    // lanes 16-31 hold row (lane-16), K slice {8..15,24..31}.
    const int row  = lane & 15;
    const int kOff = (lane >> 4) * 8;

    const float* xrow0 = x + (size_t)(tokenBase + row) * HIDDEN + kOff;
    const float* xrow1 = xrow0 + (size_t)16 * HIDDEN;
    const float* wrow  = w + (size_t)row * HIDDEN + kOff;   // expert tile t adds t*16 rows

    v8f acc0[4] = {};   // token tile 0, expert tiles 0..3
    v8f acc1[4] = {};   // token tile 1

    #pragma unroll 2
    for (int k0 = 0; k0 < HIDDEN; k0 += 32) {
        v16bf a0 = load16_cvt(xrow0 + k0);
        v16bf a1 = load16_cvt(xrow1 + k0);
        #pragma unroll
        for (int t = 0; t < 4; ++t) {
            v16bf b = load16_cvt(wrow + (size_t)(t * 16) * HIDDEN + k0);
            acc0[t] = __builtin_amdgcn_wmma_f32_16x16x32_bf16(
                false, a0, false, b, (short)0, acc0[t], false, false);
            acc1[t] = __builtin_amdgcn_wmma_f32_16x16x32_bf16(
                false, a1, false, b, (short)0, acc1[t], false, false);
        }
    }

    // C/D layout: VGPR r -> M=r (lanes 0-15) or M=8+r (lanes 16-31); lane%16 -> N.
    const int mBase = (lane >> 4) * 8;
    const int eCol  = lane & 15;
    #pragma unroll
    for (int t = 0; t < 4; ++t) {
        #pragma unroll
        for (int r = 0; r < 8; ++r) {
            sLogits[wave][mBase + r][t * 16 + eCol]      = acc0[t][r];
            sLogits[wave][16 + mBase + r][t * 16 + eCol] = acc1[t][r];
        }
    }
    __syncthreads();

    // One lane per token: softmax stats + iterative top-6 (descending, lowest-index ties).
    float* myrow = &sLogits[wave][lane][0];
    const int token = tokenBase + lane;

    float mx = -INFINITY;
    for (int e = 0; e < NEXP; ++e) mx = fmaxf(mx, myrow[e]);
    float sum = 0.0f;
    for (int e = 0; e < NEXP; ++e) sum += __expf(myrow[e] - mx);
    const float sscale = SCALE / sum;

    #pragma unroll 1
    for (int k = 0; k < TOPK; ++k) {
        float best = -INFINITY; int bi = 0;
        for (int e = 0; e < NEXP; ++e) {
            float v = myrow[e];
            if (v > best) { best = v; bi = e; }
        }
        idx_out[token * TOPK + k] = bi;
        w_out[token * TOPK + k]   = __expf(best - mx) * sscale;
        myrow[bi] = -INFINITY;
    }
}

extern "C" void kernel_launch(void* const* d_in, const int* in_sizes, int n_in,
                              void* d_out, int out_size, void* d_ws, size_t ws_size,
                              hipStream_t stream) {
    const float* x = (const float*)d_in[0];   // [4,4096,4096] f32
    const float* w = (const float*)d_in[1];   // [64,4096] f32
    int*   idx_out = (int*)d_out;                              // first 16384*6 (int32 bits)
    float* w_out   = (float*)d_out + (size_t)TOKENS * TOPK;    // next 16384*6 f32

    dim3 block(128);                 // 4 waves; 32 tokens per wave
    dim3 grid(TOKENS / 128);         // 128 blocks
    TinyTopkRouter_kernel<<<grid, block, 0, stream>>>(x, w, idx_out, w_out);
}
